// RealGatedDeltaNet_37417755083569
// MI455X (gfx1250) — matrix operand order repaired and checked by reading, not compiled, not run on hardware
//
#include <hip/hip_runtime.h>

// ---------------------------------------------------------------------------
// Gated DeltaNet forward for MI455X (gfx1250, wave32, WMMA bf16 16x16x32,
// Tensor Data Mover global->LDS staging in the dense GEMMs).
// ---------------------------------------------------------------------------

#define HID       2048
#define NKH       16
#define NVH       32
#define DK        128
#define DV        128
#define KEY_DIM   (NKH*DK)            // 2048
#define VAL_DIM   (NVH*DV)            // 4096
#define CONV_DIM  (2*KEY_DIM+VAL_DIM) // 8192
#define K_CONV    4
#define T_SEQ     2048
#define BATCH     2
#define MTOK      (BATCH*T_SEQ)       // 4096 tokens
#define CHUNK     64
#define NCHUNK    (T_SEQ/CHUNK)       // 32

typedef __bf16 bf16_t;
typedef __attribute__((ext_vector_type(16))) __bf16 bf16x16;
typedef __attribute__((ext_vector_type(8)))  __bf16 bf16x8;
typedef __attribute__((ext_vector_type(8)))  float  f32x8;
typedef __attribute__((ext_vector_type(4)))  unsigned int u32x4;
typedef __attribute__((ext_vector_type(8)))  int i32x8;

#define F8ZERO (f32x8){0.f,0.f,0.f,0.f,0.f,0.f,0.f,0.f}

static __device__ __forceinline__ f32x8 wmma_bf16(bf16x16 a, bf16x16 b, f32x8 c) {
  return __builtin_amdgcn_wmma_f32_16x16x32_bf16(false, a, false, b, (short)0, c, false, false);
}

union FragU { bf16x16 v; bf16x8 h[2]; };

// A fragment: LDS row-major [16 m][strideH], 32 K values starting at p.
// ISA layout: m = lane&15; K(j) = 8*(lane>>4) + j (j<8), 16 + 8*(lane>>4) + (j-8).
static __device__ __forceinline__ bf16x16 load_frag_a(const bf16_t* p, int strideH, int lane) {
  const bf16_t* r = p + (lane & 15)*strideH + (lane >> 4)*8;
  FragU u;
  u.h[0] = *(const bf16x8*)(r);
  u.h[1] = *(const bf16x8*)(r + 16);
  return u.v;
}
// B fragment: LDS is n-major [16 n][strideH], 32 K values starting at p.
// ISA layout: n = lane&15; K(j) = 8*(lane>>4) + j (j<8), 16 + 8*(lane>>4) + (j-8).
static __device__ __forceinline__ bf16x16 load_frag_b(const bf16_t* p, int strideH, int lane) {
  const bf16_t* r = p + (lane & 15)*strideH + (lane >> 4)*8;
  FragU u;
  u.h[0] = *(const bf16x8*)(r);
  u.h[1] = *(const bf16x8*)(r + 16);
  return u.v;
}

// ---------------------------------------------------------------------------
// Tensor Data Mover: load a 128-row x 32-col bf16 tile (row stride K elements)
// from global memory into LDS, inserting 4 DWORDs of pad every 16 DWORDs so
// the LDS image has row stride 40 halfs (matches the WMMA fragment loaders).
// D# packing per CDNA5 ISA 8.3/8.4: group0 {count=1, lds_addr, global_addr,
// type=2}; group1 {data_size=1(2B), pad_enable, pad_interval=3(16dw),
// pad_amount=3(4dw), tensor_dim0=K, tensor_dim1=1<<20, tile=32x128,
// dim0_stride=K}.  Issued by one wave; tracked by TENSORcnt.
// ---------------------------------------------------------------------------
static __device__ __forceinline__ unsigned lds_off(const void* p) {
  return (unsigned)(__SIZE_TYPE__)(__attribute__((address_space(3))) const void*)p;
}

static __device__ __forceinline__ void tdm_load_tile(unsigned lds_addr, const bf16_t* gsrc,
                                                     unsigned Kelems) {
  unsigned long long ga = (unsigned long long)(__SIZE_TYPE__)gsrc;
  u32x4 g0;
  g0.x = 1u;                                                  // count=1, user mode
  g0.y = lds_addr;                                            // LDS byte address
  g0.z = (unsigned)ga;                                        // global_addr[31:0]
  g0.w = (unsigned)((ga >> 32) & 0x01FFFFFFull) | (2u << 30); // addr[56:32] | type=2
  i32x8 g1;
  g1[0] = (1 << 16) | (1 << 20) | (3 << 22) | (3 << 25);      // 2B, pad 4dw/16dw
  g1[1] = (int)((Kelems & 0xFFFFu) << 16);                    // tensor_dim0[15:0]
  g1[2] = (int)(Kelems >> 16);                                // tensor_dim0[31:16] | tdim1 lo=0
  g1[3] = (int)(16u | (32u << 16));                           // tdim1 hi (1<<20) | tile_dim0=32
  g1[4] = 128;                                                // tile_dim1=128, tile_dim2=0
  g1[5] = (int)Kelems;                                        // dim0_stride[31:0]
  g1[6] = 0;                                                  // dim0_stride hi | dim1_stride lo
  g1[7] = 0;
  asm volatile("tensor_load_to_lds %0, %1" :: "s"(g0), "s"(g1) : "memory");
}

// ---------------------------------------------------------------------------
// f32 -> bf16 convert
// ---------------------------------------------------------------------------
__global__ void cvt_bf16_kernel(const float* __restrict__ src, bf16_t* __restrict__ dst, size_t n) {
  size_t i = (size_t)blockIdx.x*256 + threadIdx.x;
  size_t stride = (size_t)gridDim.x*256;
  for (; i < n; i += stride) dst[i] = (bf16_t)src[i];
}

// ---------------------------------------------------------------------------
// GEMM: C[M,N] f32 = A[M,K]bf16 (row major) * B[N,K]bf16 (row major, i.e. Bᵀ)
// Block tile 128x128, K-step 32, 8 waves (wave w owns 16-row strip x 128 col).
// Tiles staged by TDM (tensor_load_to_lds) with double buffering: the DMA of
// K-slice i+1 overlaps the 8-WMMA compute of slice i.
// ---------------------------------------------------------------------------
__global__ __launch_bounds__(256) void gemm_bf16_nt_kernel(
    const bf16_t* __restrict__ A, const bf16_t* __restrict__ B,
    float* __restrict__ C, int M, int N, int K)
{
  __shared__ bf16_t sA[2][128*40];
  __shared__ bf16_t sB[2][128*40];
  int tid = threadIdx.x, lane = tid & 31, w = tid >> 5;
  int m0 = blockIdx.y*128, n0 = blockIdx.x*128;
  const bf16_t* Abase = A + (size_t)m0*K;
  const bf16_t* Bbase = B + (size_t)n0*K;
  f32x8 acc[8];
#pragma unroll
  for (int i = 0; i < 8; ++i) acc[i] = F8ZERO;

  if (tid < 32) {                       // wave 0 drives the TDM
    tdm_load_tile(lds_off(&sA[0][0]), Abase, (unsigned)K);
    tdm_load_tile(lds_off(&sB[0][0]), Bbase, (unsigned)K);
  }

  int ib = 0;
  for (int k0 = 0; k0 < K; k0 += 32, ib ^= 1) {
    __builtin_amdgcn_s_wait_tensorcnt(0);   // wave0: current tiles landed
    __syncthreads();                        // publish to all waves
    if ((k0 + 32 < K) && tid < 32) {        // prefetch next slice into other buffer
      tdm_load_tile(lds_off(&sA[ib ^ 1][0]), Abase + (k0 + 32), (unsigned)K);
      tdm_load_tile(lds_off(&sB[ib ^ 1][0]), Bbase + (k0 + 32), (unsigned)K);
    }
    bf16x16 af = load_frag_a(&sA[ib][w*16*40], 40, lane);
#pragma unroll
    for (int nt = 0; nt < 8; ++nt) {
      bf16x16 bfr = load_frag_b(&sB[ib][nt*16*40], 40, lane);
      acc[nt] = wmma_bf16(af, bfr, acc[nt]);
    }
    __syncthreads();                        // compute done before buffer reuse
  }

  int mr = m0 + w*16 + 8*(lane >> 4);
#pragma unroll
  for (int nt = 0; nt < 8; ++nt) {
    int nn = n0 + nt*16 + (lane & 15);
#pragma unroll
    for (int r = 0; r < 8; ++r)
      C[(size_t)(mr + r)*N + nn] = acc[nt][r];
  }
}

// ---------------------------------------------------------------------------
// a/b projections (N=32 each) -> g, beta.   One block per token, 64 threads.
// ---------------------------------------------------------------------------
__global__ __launch_bounds__(64) void ab_kernel(
    const float* __restrict__ hs, const float* __restrict__ Wa, const float* __restrict__ Wb,
    const float* __restrict__ A_log, const float* __restrict__ dt_bias,
    float* __restrict__ g, float* __restrict__ beta)
{
  int m = blockIdx.x, tid = threadIdx.x;
  __shared__ float sh[64];
  const float* Wrow = (tid < 32) ? (Wa + (size_t)tid*HID) : (Wb + (size_t)(tid - 32)*HID);
  float acc = 0.f;
  for (int kk = 0; kk < HID; kk += 64) {
    sh[tid] = hs[(size_t)m*HID + kk + tid];
    __syncthreads();
#pragma unroll 8
    for (int j = 0; j < 64; ++j) acc += sh[j]*Wrow[kk + j];
    __syncthreads();
  }
  if (tid < 32) {
    float x = acc + dt_bias[tid];
    float sp = (x > 20.f) ? x : log1pf(__expf(x));
    g[(size_t)m*NVH + tid] = -__expf(A_log[tid])*sp;
  } else {
    beta[(size_t)m*NVH + (tid - 32)] = 1.f/(1.f + __expf(-acc));
  }
}

// ---------------------------------------------------------------------------
// Depthwise causal conv (K=4) + SiLU.  mixed f32 [B,T,C] -> xact bf16 [B,T,C]
// ---------------------------------------------------------------------------
__global__ void conv_silu_kernel(const float* __restrict__ mixed, const float* __restrict__ w,
                                 bf16_t* __restrict__ xact)
{
  size_t idx = (size_t)blockIdx.x*256 + threadIdx.x;   // < B*T*CONV_DIM
  int c = (int)(idx & (CONV_DIM - 1));
  size_t bt = idx >> 13;                               // /8192
  int t = (int)(bt & (T_SEQ - 1));
  float acc = 0.f;
#pragma unroll
  for (int j = 0; j < K_CONV; ++j) {
    int tt = t - (K_CONV - 1) + j;
    if (tt >= 0) acc += w[c*K_CONV + j]*mixed[idx + (size_t)((long)(j - (K_CONV - 1)))*CONV_DIM];
  }
  float s = acc/(1.f + __expf(-acc));
  xact[idx] = (bf16_t)s;
}

__global__ void conv_state_kernel(const float* __restrict__ mixed, float* __restrict__ cs) {
  int idx = blockIdx.x*256 + threadIdx.x;              // < B*CONV_DIM*3
  if (idx >= BATCH*CONV_DIM*(K_CONV - 1)) return;
  int j = idx % 3;
  int c = (idx/3) % CONV_DIM;
  int b = idx/(3*CONV_DIM);
  cs[idx] = mixed[((size_t)b*T_SEQ + (T_SEQ - 3 + j))*CONV_DIM + c];
}

// ---------------------------------------------------------------------------
// q/k L2-norm (+ q * Dk^-1/2), relayout head-major; v relayout. 128 thr/blk.
// ---------------------------------------------------------------------------
__global__ __launch_bounds__(128) void prep_qkv_kernel(
    const bf16_t* __restrict__ xact, bf16_t* __restrict__ qh,
    bf16_t* __restrict__ kh, bf16_t* __restrict__ vh)
{
  int idx = blockIdx.x;              // b*T*64
  int head = idx & 63;
  int bt = idx >> 6;
  int b = bt >> 11;                  // / T_SEQ
  int t = bt & (T_SEQ - 1);
  int d = threadIdx.x;
  const bf16_t* base = xact + (size_t)bt*CONV_DIM;
  __shared__ float red[128];
  if (head < 32) {
    int isK = head >= 16;
    int hh = head & 15;
    float x = (float)base[(isK ? KEY_DIM : 0) + hh*DK + d];
    red[d] = x*x;
    __syncthreads();
    for (int s = 64; s > 0; s >>= 1) { if (d < s) red[d] += red[d + s]; __syncthreads(); }
    float scale = rsqrtf(red[0] + 1e-6f)*(isK ? 1.0f : 0.08838834764831845f);
    bf16_t* dst = isK ? kh : qh;
    dst[(((size_t)b*NKH + hh)*T_SEQ + t)*DK + d] = (bf16_t)(x*scale);
  } else {
    int hh = head - 32;
    vh[(((size_t)b*NVH + hh)*T_SEQ + t)*DV + d] = base[2*KEY_DIM + hh*DV + d];
  }
}

// ---------------------------------------------------------------------------
// Chunked gated delta rule.  One block per (b,h); 8 waves; 286 KB dynamic LDS.
// ---------------------------------------------------------------------------
#define DSM_S    0        // f32  [128][128]          65536
#define DSM_ST   65536    // bf16 [128][136]  (S^T)   34816
#define DSM_K    100352   // bf16 [64][136]           17408
#define DSM_Q    117760   // bf16 [64][136]           17408
#define DSM_QG   135168   // bf16 [64][136]           17408
#define DSM_KB   152576   // bf16 [64][136] / KDT[128][72]  18432
#define DSM_KBGT 171008   // bf16 [128][72]           18432
#define DSM_VBT  189440   // bf16 [128][72] (vbeta^T then vnew^T) 18432
#define DSM_AF   207872   // f32  [64][66] / attn bf16 [64][72]   16896
#define DSM_TM   224768   // f32  [64][66]            16896
#define DSM_TMB  241664   // bf16 [64][72]             9216
#define DSM_KC   250880   // bf16 [64][136]           17408
#define DSM_DEC  268288   // f32  [64][66]            16896
#define DSM_GC   285184
#define DSM_EG   285440
#define DSM_BET  285696
#define DSM_TOTAL 285952

__global__ __launch_bounds__(256) void delta_kernel(
    const bf16_t* __restrict__ qh, const bf16_t* __restrict__ kh, const bf16_t* __restrict__ vh,
    const float* __restrict__ gbuf, const float* __restrict__ betab,
    float* __restrict__ core, float* __restrict__ state_out)
{
  extern __shared__ char smem[];
  float*  Sm   = (float*) (smem + DSM_S);
  bf16_t* STb  = (bf16_t*)(smem + DSM_ST);
  bf16_t* Kt   = (bf16_t*)(smem + DSM_K);
  bf16_t* Qt   = (bf16_t*)(smem + DSM_Q);
  bf16_t* QG   = (bf16_t*)(smem + DSM_QG);
  bf16_t* KB   = (bf16_t*)(smem + DSM_KB);
  bf16_t* KBGT = (bf16_t*)(smem + DSM_KBGT);
  bf16_t* VBT  = (bf16_t*)(smem + DSM_VBT);
  float*  Af   = (float*) (smem + DSM_AF);
  float*  Tm32 = (float*) (smem + DSM_TM);
  bf16_t* Tmb  = (bf16_t*)(smem + DSM_TMB);
  bf16_t* KC   = (bf16_t*)(smem + DSM_KC);
  float*  DEC  = (float*) (smem + DSM_DEC);
  float*  gcv  = (float*) (smem + DSM_GC);
  float*  egv  = (float*) (smem + DSM_EG);
  float*  betv = (float*) (smem + DSM_BET);

  int b = blockIdx.x >> 5, h = blockIdx.x & 31, hk = h >> 1;
  const bf16_t* qp = qh + ((size_t)(b*NKH + hk))*T_SEQ*DK;
  const bf16_t* kp = kh + ((size_t)(b*NKH + hk))*T_SEQ*DK;
  const bf16_t* vp = vh + ((size_t)(b*NVH + h ))*T_SEQ*DV;
  int tid = threadIdx.x, lane = tid & 31, w = tid >> 5;

  for (int i = tid; i < 128*128; i += 256) Sm[i] = 0.f;
  for (int i = tid; i < 128*136; i += 256) STb[i] = (bf16_t)0.f;
  __syncthreads();

  for (int nc = 0; nc < NCHUNK; ++nc) {
    int t0 = nc*CHUNK;
    if (tid < 64) {
      size_t gi = ((size_t)b*T_SEQ + t0 + tid)*NVH + h;
      gcv[tid]  = gbuf[gi];
      betv[tid] = betab[gi];
    }
    __syncthreads();
    if (tid == 0) { float s = 0.f; for (int t = 0; t < 64; ++t) { s += gcv[t]; gcv[t] = s; } }
    __syncthreads();
    if (tid < 64) egv[tid] = __expf(gcv[tid]);
    __syncthreads();
    float glast = gcv[63];

    {   // load q/k/v tiles + derived operands
      int r = tid >> 2, cs = (tid & 3)*32;
      const bf16_t* qrow = qp + (size_t)(t0 + r)*DK;
      const bf16_t* krow = kp + (size_t)(t0 + r)*DK;
      const bf16_t* vrow = vp + (size_t)(t0 + r)*DV;
      float bet = betv[r], eg = egv[r];
      for (int c = cs; c < cs + 32; ++c) {
        float kx = (float)krow[c];
        Kt[r*136 + c]  = (bf16_t)kx;
        KB[r*136 + c]  = (bf16_t)(kx*bet);
        KBGT[c*72 + r] = (bf16_t)(kx*bet*eg);
        float qx = (float)qrow[c];
        Qt[r*136 + c]  = (bf16_t)qx;
        QG[r*136 + c]  = (bf16_t)(qx*eg);
        float vx = (float)vrow[c];
        VBT[c*72 + r]  = (bf16_t)(vx*bet);
      }
    }
    for (int e = tid; e < 4096; e += 256) {
      int i = e >> 6, j = e & 63;
      DEC[i*66 + j] = (j <= i) ? __expf(gcv[i] - gcv[j]) : 0.f;
    }
    __syncthreads();

    // A = strict_lower( -(k_beta @ k^T) * decay )
    for (int tile = w; tile < 16; tile += 8) {
      int ti = tile >> 2, tj = tile & 3;
      f32x8 acc = F8ZERO;
#pragma unroll
      for (int kk = 0; kk < 4; ++kk)
        acc = wmma_bf16(load_frag_a(KB + ti*16*136 + kk*32, 136, lane),
                        load_frag_b(Kt + tj*16*136 + kk*32, 136, lane), acc);
#pragma unroll
      for (int r = 0; r < 8; ++r) {
        int i = ti*16 + r + 8*(lane >> 4), j = tj*16 + (lane & 15);
        Af[i*66 + j] = (i > j) ? (-acc[r]*DEC[i*66 + j]) : 0.f;
      }
    }
    __syncthreads();

    // Tm = (I - A)^-1 by forward substitution; one column per thread, no deps.
    if (tid < 64) {
      int c = tid;
      for (int i = 0; i < 64; ++i) {
        float s = (i == c) ? 1.f : 0.f;
        for (int j = 0; j < i; ++j) s += Af[i*66 + j]*Tm32[j*66 + c];
        Tm32[i*66 + c] = s;
      }
    }
    __syncthreads();
    for (int e = tid; e < 4096; e += 256)
      Tmb[(e >> 6)*72 + (e & 63)] = (bf16_t)Tm32[(e >> 6)*66 + (e & 63)];
    __syncthreads();

    // KC = -(Tm @ (k_beta * exp(g)))   (negated so v_new accumulates with +)
    for (int tile = w; tile < 32; tile += 8) {
      int ti = tile >> 3, tj = tile & 7;
      f32x8 acc = F8ZERO;
#pragma unroll
      for (int kk = 0; kk < 2; ++kk)
        acc = wmma_bf16(load_frag_a(Tmb + ti*16*72 + kk*32, 72, lane),
                        load_frag_b(KBGT + tj*16*72 + kk*32, 72, lane), acc);
#pragma unroll
      for (int r = 0; r < 8; ++r)
        KC[(ti*16 + r + 8*(lane >> 4))*136 + tj*16 + (lane & 15)] = (bf16_t)(-acc[r]);
    }
    __syncthreads();

    // v_new = Tm@v_beta + KC@S ; write v_new^T over VBT
    {
      int rg = w >> 1, c0 = (w & 1)*4;
      f32x8 vacc[4];
      for (int q4 = 0; q4 < 4; ++q4) {
        int tj = c0 + q4;
        f32x8 acc = F8ZERO;
#pragma unroll
        for (int kk = 0; kk < 2; ++kk)
          acc = wmma_bf16(load_frag_a(Tmb + rg*16*72 + kk*32, 72, lane),
                          load_frag_b(VBT + tj*16*72 + kk*32, 72, lane), acc);
#pragma unroll
        for (int kk = 0; kk < 4; ++kk)
          acc = wmma_bf16(load_frag_a(KC + rg*16*136 + kk*32, 136, lane),
                          load_frag_b(STb + tj*16*136 + kk*32, 136, lane), acc);
        vacc[q4] = acc;
      }
      __syncthreads();
      for (int q4 = 0; q4 < 4; ++q4) {
        int tj = c0 + q4;
#pragma unroll
        for (int r = 0; r < 8; ++r) {
          int m = rg*16 + r + 8*(lane >> 4), n = tj*16 + (lane & 15);
          VBT[n*72 + m] = (bf16_t)vacc[q4][r];
        }
      }
    }
    __syncthreads();

    // attn = tril( (q @ k^T) * decay )   (bf16, reuses Af region)
    bf16_t* Ab = (bf16_t*)Af;
    for (int tile = w; tile < 16; tile += 8) {
      int ti = tile >> 2, tj = tile & 3;
      f32x8 acc = F8ZERO;
#pragma unroll
      for (int kk = 0; kk < 4; ++kk)
        acc = wmma_bf16(load_frag_a(Qt + ti*16*136 + kk*32, 136, lane),
                        load_frag_b(Kt + tj*16*136 + kk*32, 136, lane), acc);
#pragma unroll
      for (int r = 0; r < 8; ++r) {
        int i = ti*16 + r + 8*(lane >> 4), j = tj*16 + (lane & 15);
        Ab[i*72 + j] = (bf16_t)((j <= i) ? acc[r]*DEC[i*66 + j] : 0.f);
      }
    }
    __syncthreads();

    // out = (q*exp(g))@S + attn@v_new  -> core (B,T,NVH,DV)
    {
      int rg = w >> 1, c0 = (w & 1)*4;
      for (int q4 = 0; q4 < 4; ++q4) {
        int tj = c0 + q4;
        f32x8 acc = F8ZERO;
#pragma unroll
        for (int kk = 0; kk < 4; ++kk)
          acc = wmma_bf16(load_frag_a(QG + rg*16*136 + kk*32, 136, lane),
                          load_frag_b(STb + tj*16*136 + kk*32, 136, lane), acc);
#pragma unroll
        for (int kk = 0; kk < 2; ++kk)
          acc = wmma_bf16(load_frag_a(Ab + rg*16*72 + kk*32, 72, lane),
                          load_frag_b(VBT + tj*16*72 + kk*32, 72, lane), acc);
#pragma unroll
        for (int r = 0; r < 8; ++r) {
          int m = rg*16 + r + 8*(lane >> 4), n = tj*16 + (lane & 15);
          core[(((size_t)b*T_SEQ + t0 + m)*NVH + h)*DV + n] = acc[r];
        }
      }
    }

    // KDT[dk][t] = k[t][dk]*exp(g_last - g_t)   (reuses KB region)
    {
      int r = tid >> 2, cs = (tid & 3)*32;
      float dk_ = __expf(glast - gcv[r]);
      bf16_t* KDT = KB;
      for (int c = cs; c < cs + 32; ++c)
        KDT[c*72 + r] = (bf16_t)((float)Kt[r*136 + c]*dk_);
    }
    __syncthreads();

    // S = S*exp(g_last) + KDT @ v_new ; refresh f32 master + bf16 S^T
    {
      float egl = __expf(glast);
      bf16_t* KDT = KB;
      for (int tj = 0; tj < 8; ++tj) {
        f32x8 acc = F8ZERO;
#pragma unroll
        for (int kk = 0; kk < 2; ++kk)
          acc = wmma_bf16(load_frag_a(KDT + w*16*72 + kk*32, 72, lane),
                          load_frag_b(VBT + tj*16*72 + kk*32, 72, lane), acc);
#pragma unroll
        for (int r = 0; r < 8; ++r) {
          int m = w*16 + r + 8*(lane >> 4), n = tj*16 + (lane & 15);
          float sv = Sm[m*128 + n]*egl + acc[r];
          Sm[m*128 + n] = sv;
          STb[n*136 + m] = (bf16_t)sv;
        }
      }
    }
    __syncthreads();
  }

  float* so = state_out + (size_t)blockIdx.x*(DK*DV);
  for (int i = tid; i < DK*DV; i += 256) so[i] = Sm[i];
}

// ---------------------------------------------------------------------------
// RMS-norm over DV + SiLU(z) gate -> bf16 (input of final GEMM)
// ---------------------------------------------------------------------------
__global__ __launch_bounds__(128) void norm_gate_kernel(
    const float* __restrict__ core, const float* __restrict__ zf,
    const float* __restrict__ norm_w, bf16_t* __restrict__ hg)
{
  int row = blockIdx.x;           // (b*T + t)*NVH + h
  int d = threadIdx.x;
  float v = core[(size_t)row*DV + d];
  __shared__ float red[128];
  red[d] = v*v;
  __syncthreads();
  for (int s = 64; s > 0; s >>= 1) { if (d < s) red[d] += red[d + s]; __syncthreads(); }
  float var = red[0]*(1.f/128.f);
  float y = norm_w[d]*v*rsqrtf(var + 1e-6f);
  float zv = zf[(size_t)row*DV + d];
  y *= zv/(1.f + __expf(-zv));
  hg[(size_t)row*DV + d] = (bf16_t)y;
}

// ---------------------------------------------------------------------------
// host launcher
// ---------------------------------------------------------------------------
extern "C" void kernel_launch(void* const* d_in, const int* in_sizes, int n_in,
                              void* d_out, int out_size, void* d_ws, size_t ws_size,
                              hipStream_t stream)
{
  (void)in_sizes; (void)n_in; (void)out_size; (void)ws_size;
  const float* hs       = (const float*)d_in[0];
  const float* W_qkv    = (const float*)d_in[1];
  const float* W_z      = (const float*)d_in[2];
  const float* W_a      = (const float*)d_in[3];
  const float* W_b      = (const float*)d_in[4];
  const float* conv_w   = (const float*)d_in[5];
  const float* A_log    = (const float*)d_in[6];
  const float* dt_bias  = (const float*)d_in[7];
  const float* norm_w   = (const float*)d_in[8];
  const float* W_out    = (const float*)d_in[9];

  float* out       = (float*)d_out;                         // [B,T,HID]
  float* state_out = out + (size_t)MTOK*HID;                // [B,NVH,DK,DV]
  float* cs_out    = state_out + (size_t)BATCH*NVH*DK*DV;   // [B,CONV_DIM,3]

  char* ws = (char*)d_ws;
  size_t o = 0;
  bf16_t* hsb    = (bf16_t*)(ws + o); o += (size_t)MTOK*HID*2;          // 16 MB
  bf16_t* wqkvb  = (bf16_t*)(ws + o); o += (size_t)CONV_DIM*HID*2;      // 32 MB
  bf16_t* wzb    = (bf16_t*)(ws + o); o += (size_t)VAL_DIM*HID*2;       // 16 MB
  bf16_t* woutb  = (bf16_t*)(ws + o); o += (size_t)HID*VAL_DIM*2;       // 16 MB
  float*  mixedf = (float*) (ws + o); o += (size_t)MTOK*CONV_DIM*4;     // 128 MB (core aliases)
  float*  zf     = (float*) (ws + o); o += (size_t)MTOK*VAL_DIM*4;      // 64 MB
  bf16_t* xactb  = (bf16_t*)(ws + o); o += (size_t)MTOK*CONV_DIM*2;     // 64 MB (hgated aliases)
  bf16_t* qhb    = (bf16_t*)(ws + o); o += (size_t)BATCH*NKH*T_SEQ*DK*2;
  bf16_t* khb    = (bf16_t*)(ws + o); o += (size_t)BATCH*NKH*T_SEQ*DK*2;
  bf16_t* vhb    = (bf16_t*)(ws + o); o += (size_t)BATCH*NVH*T_SEQ*DV*2;
  float*  gbuf   = (float*) (ws + o); o += (size_t)MTOK*NVH*4;
  float*  betab  = (float*) (ws + o); o += (size_t)MTOK*NVH*4;
  float*  core   = mixedf;                 // alias: mixed dead after conv/conv_state
  bf16_t* hgated = xactb;                  // alias: xact dead after prep_qkv

  // f32 -> bf16 conversions
  cvt_bf16_kernel<<<4096, 256, 0, stream>>>(hs,    hsb,   (size_t)MTOK*HID);
  cvt_bf16_kernel<<<8192, 256, 0, stream>>>(W_qkv, wqkvb, (size_t)CONV_DIM*HID);
  cvt_bf16_kernel<<<4096, 256, 0, stream>>>(W_z,   wzb,   (size_t)VAL_DIM*HID);
  cvt_bf16_kernel<<<4096, 256, 0, stream>>>(W_out, woutb, (size_t)HID*VAL_DIM);

  // mixed = hs @ W_qkv^T ; z = hs @ W_z^T
  gemm_bf16_nt_kernel<<<dim3(CONV_DIM/128, MTOK/128), 256, 0, stream>>>(
      hsb, wqkvb, mixedf, MTOK, CONV_DIM, HID);
  gemm_bf16_nt_kernel<<<dim3(VAL_DIM/128, MTOK/128), 256, 0, stream>>>(
      hsb, wzb, zf, MTOK, VAL_DIM, HID);

  // g, beta
  ab_kernel<<<MTOK, 64, 0, stream>>>(hs, W_a, W_b, A_log, dt_bias, gbuf, betab);

  // conv state (reads mixed f32, before core aliases over it)
  conv_state_kernel<<<(BATCH*CONV_DIM*3 + 255)/256, 256, 0, stream>>>(mixedf, cs_out);

  // depthwise conv + SiLU
  conv_silu_kernel<<<((size_t)MTOK*CONV_DIM)/256, 256, 0, stream>>>(mixedf, conv_w, xactb);

  // q/k L2 norm + head-major relayout, v relayout
  prep_qkv_kernel<<<MTOK*64, 128, 0, stream>>>(xactb, qhb, khb, vhb);

  // chunked gated delta rule (writes core + final state)
  (void)hipFuncSetAttribute(reinterpret_cast<const void*>(delta_kernel),
                            hipFuncAttributeMaxDynamicSharedMemorySize, DSM_TOTAL);
  delta_kernel<<<BATCH*NVH, 256, DSM_TOTAL, stream>>>(qhb, khb, vhb, gbuf, betab,
                                                      core, state_out);

  // RMS norm + SiLU(z) gate
  norm_gate_kernel<<<MTOK*NVH, 128, 0, stream>>>(core, zf, norm_w, hgated);

  // out = hgated @ W_out^T
  gemm_bf16_nt_kernel<<<dim3(HID/128, MTOK/128), 256, 0, stream>>>(
      hgated, woutb, out, MTOK, HID, VAL_DIM);
}
